// ImprovedLoss_13005160972365
// MI455X (gfx1250) — compile-verified, matrix operand order checked by reading, not run on hardware
//
#include <hip/hip_runtime.h>
#include <math.h>

// ---------------------------------------------------------------------------
// Types for WMMA operands
// ---------------------------------------------------------------------------
typedef __attribute__((ext_vector_type(2)))  float    v2f;
typedef __attribute__((ext_vector_type(8)))  float    v8f;
typedef __attribute__((ext_vector_type(16))) _Float16 v16h;

#if defined(__AMDGCN__) && __has_builtin(__builtin_amdgcn_wmma_f32_16x16x4_f32)
#define PMW_F32_WMMA 1
#endif

// ---------------------------------------------------------------------------
// Pairwise min kernel: for each point in X (nX x 3, per batch), the minimum
// squared distance to any point in Y (nY x 3).  One wave owns a 16-row tile of
// X; each loop iteration computes a 16x16 tile of inner products with a single
// v_wmma_f32_16x16x4_f32 (d=3 padded to K=4), applies |x|^2+|y|^2-2ab, and
// keeps a running min.  Cross-lane min via shfl_xor within 16-lane halves.
// ---------------------------------------------------------------------------
#define PMW_WAVES 8

__global__ void pairwise_min_wmma(const float* __restrict__ X,
                                  const float* __restrict__ Y,
                                  int nX, int nY,
                                  float* __restrict__ outMin) {
  const int batch = blockIdx.y;
  const float* Xb = X + (size_t)batch * nX * 3;
  const float* Yb = Y + (size_t)batch * nY * 3;

  const int wave = threadIdx.x >> 5;
  const int lane = threadIdx.x & 31;
  const int half = lane >> 4;        // 0: lanes 0-15, 1: lanes 16-31
  const int r    = lane & 15;
  const int tile = blockIdx.x * PMW_WAVES + wave;
  const int r0   = tile * 16;
  if (r0 >= nX) return;              // wave-uniform

  const int row = r0 + r;
  const float x0 = Xb[row * 3 + 0];
  const float x1 = Xb[row * 3 + 1];
  const float x2 = Xb[row * 3 + 2];
  const float xn = x0 * x0 + x1 * x1 + x2 * x2;

  // per-lane copies of |x_m|^2 for the rows this half touches (M=v or v+8)
  float xnv[8];
#pragma unroll
  for (int v = 0; v < 8; ++v) xnv[v] = __shfl(xn, v + half * 8, 32);

#if PMW_F32_WMMA
  // A 16x4 f32: VGPR0 = K0 (lo half) / K2 (hi half); VGPR1 = K1 / K3(pad=0)
  v2f a;
  a.x = half ? x2 : x0;
  a.y = half ? 0.0f : x1;
#else
  // A 16x32 f16: lanes 0-15 hold K0..7 in elements 0..7, upper lanes K8..15=0
  v16h a = {};
  if (!half) { a[0] = (_Float16)x0; a[1] = (_Float16)x1; a[2] = (_Float16)x2; }
#endif

  float minD[8];
#pragma unroll
  for (int v = 0; v < 8; ++v) minD[v] = 3.0e38f;

  for (int n0 = 0; n0 < nY; n0 += 16) {
    const int col = n0 + r;
    const float y0 = Yb[col * 3 + 0];
    const float y1 = Yb[col * 3 + 1];
    const float y2 = Yb[col * 3 + 2];
    const float yn = y0 * y0 + y1 * y1 + y2 * y2;

    v8f c = {};
#if PMW_F32_WMMA
    v2f bmat;
    bmat.x = half ? y2 : y0;       // rows K=0 (lo) / K=2 (hi)
    bmat.y = half ? 0.0f : y1;     // rows K=1 (lo) / K=3 pad (hi)
    c = __builtin_amdgcn_wmma_f32_16x16x4_f32(
        false, a, false, bmat, (short)0, c, false, false);
#else
    v16h bmat = {};
    if (!half) { bmat[0] = (_Float16)y0; bmat[1] = (_Float16)y1; bmat[2] = (_Float16)y2; }
    c = __builtin_amdgcn_wmma_f32_16x16x32_f16(
        false, a, false, bmat, (short)0, c, false, false);
#endif

#pragma unroll
    for (int v = 0; v < 8; ++v) {
      float sq = xnv[v] + yn - 2.0f * c[v];
      sq = sq > 0.0f ? sq : 0.0f;
      minD[v] = fminf(minD[v], sq);
    }
  }

  // min across the 16 lanes of each half (each lane held one column N=r)
#pragma unroll
  for (int v = 0; v < 8; ++v) {
    float m = minD[v];
    m = fminf(m, __shfl_xor(m, 1, 32));
    m = fminf(m, __shfl_xor(m, 2, 32));
    m = fminf(m, __shfl_xor(m, 4, 32));
    m = fminf(m, __shfl_xor(m, 8, 32));
    minD[v] = m;
  }
  if (r == 0) {  // lanes 0 (rows 0..7) and 16 (rows 8..15)
#pragma unroll
    for (int v = 0; v < 8; ++v)
      outMin[(size_t)batch * nX + r0 + half * 8 + v] = minD[v];
  }
}

// ---------------------------------------------------------------------------
// Single-block array reduction. mode 0: sum(x); mode 1: sum(safe_sqrt(x)).
// Writes out[0] = sum * scale.  Deterministic (no atomics).
// ---------------------------------------------------------------------------
__global__ void reduce_sum_kernel(const float* __restrict__ a, int n, int mode,
                                  float scale, float* __restrict__ out) {
  __shared__ float s[1024];
  float acc = 0.0f;
  for (int i = threadIdx.x; i < n; i += blockDim.x) {
    float v = a[i];
    if (mode == 1) v = v > 0.0f ? sqrtf(v) : 0.0f;
    acc += v;
  }
  s[threadIdx.x] = acc;
  __syncthreads();
  for (int o = blockDim.x >> 1; o > 0; o >>= 1) {
    if (threadIdx.x < (unsigned)o) s[threadIdx.x] += s[threadIdx.x + o];
    __syncthreads();
  }
  if (threadIdx.x == 0) out[0] = s[0] * scale;
}

// ---------------------------------------------------------------------------
// EMD: in-LDS bitonic sort of one column (n = 8192 values, stride-3 gather).
// ---------------------------------------------------------------------------
__global__ void bitonic_sort_col(const float* __restrict__ src, int n, int col,
                                 float* __restrict__ dst) {
  __shared__ float s[8192];
  for (int i = threadIdx.x; i < n; i += blockDim.x) s[i] = src[i * 3 + col];
  __syncthreads();
  for (int k = 2; k <= n; k <<= 1) {
    for (int j = k >> 1; j > 0; j >>= 1) {
      for (int i = threadIdx.x; i < n; i += blockDim.x) {
        int ixj = i ^ j;
        if (ixj > i) {
          bool up = ((i & k) == 0);
          float va = s[i], vb = s[ixj];
          if ((va > vb) == up) { s[i] = vb; s[ixj] = va; }
        }
      }
      __syncthreads();
    }
  }
  for (int i = threadIdx.x; i < n; i += blockDim.x) dst[i] = s[i];
}

__global__ void emd_reduce_kernel(const float* __restrict__ sortBuf, int n,
                                  float* __restrict__ out) {
  __shared__ float s[1024];
  float acc = 0.0f;
  for (int i = threadIdx.x; i < 3 * n; i += blockDim.x) {
    int c = i / n, k = i % n;
    float d = sortBuf[c * n + k] - sortBuf[(3 + c) * n + k];
    acc += d * d;
  }
  s[threadIdx.x] = acc;
  __syncthreads();
  for (int o = blockDim.x >> 1; o > 0; o >>= 1) {
    if (threadIdx.x < (unsigned)o) s[threadIdx.x] += s[threadIdx.x + o];
    __syncthreads();
  }
  if (threadIdx.x == 0) out[0] = s[0] * (0.1f / (3.0f * (float)n));
}

// ---------------------------------------------------------------------------
// Uniformity: per-row top-50 nearest distances over the 8192 flattened points,
// then sample std (ddof=1) of entries 1..49 (entry 0 is the self-distance 0).
// ---------------------------------------------------------------------------
__global__ void uniformity_rows_kernel(const float* __restrict__ P, int n,
                                       float* __restrict__ rowStd) {
  const int row = blockIdx.x * blockDim.x + threadIdx.x;
  if (row >= n) return;
  const float px = P[row * 3 + 0];
  const float py = P[row * 3 + 1];
  const float pz = P[row * 3 + 2];

  float knn[50];
#pragma unroll
  for (int i = 0; i < 50; ++i) knn[i] = 3.0e38f;

  for (int j = 0; j < n; ++j) {
    float dx = px - P[j * 3 + 0];
    float dy = py - P[j * 3 + 1];
    float dz = pz - P[j * 3 + 2];
    float sq = dx * dx + dy * dy + dz * dz;
    float d  = sq > 0.0f ? sqrtf(sq) : 0.0f;
    if (d < knn[49]) {
      int p = 49;
      while (p > 0 && knn[p - 1] > d) { knn[p] = knn[p - 1]; --p; }
      knn[p] = d;
    }
  }

  float mean = 0.0f;
  for (int i = 1; i < 50; ++i) mean += knn[i];
  mean *= (1.0f / 49.0f);
  float var = 0.0f;
  for (int i = 1; i < 50; ++i) {
    float df = knn[i] - mean;
    var += df * df;
  }
  var *= (1.0f / 48.0f);  // ddof=1
  rowStd[row] = sqrtf(var);
}

// ---------------------------------------------------------------------------
// Spread: relu(0.5 - mean_{b,d} std(pred[b,:,d], ddof=1)) * 10
// ---------------------------------------------------------------------------
__global__ void spread_kernel(const float* __restrict__ pred, int B, int N,
                              float* __restrict__ out) {
  __shared__ float sstd[8];
  const int t = threadIdx.x;
  if (t < B * 3) {
    const int b = t / 3, d = t % 3;
    const float* p = pred + (size_t)b * N * 3 + d;
    float mean = 0.0f;
    for (int i = 0; i < N; ++i) mean += p[i * 3];
    mean /= (float)N;
    float var = 0.0f;
    for (int i = 0; i < N; ++i) {
      float df = p[i * 3] - mean;
      var += df * df;
    }
    var /= (float)(N - 1);
    sstd[t] = sqrtf(var);
  }
  __syncthreads();
  if (t == 0) {
    float m = 0.0f;
    for (int i = 0; i < B * 3; ++i) m += sstd[i];
    m /= (float)(B * 3);
    float sp = 0.5f - m;
    out[0] = (sp > 0.0f ? sp : 0.0f) * 10.0f;
  }
}

// ---------------------------------------------------------------------------
// Repulsion: 100x100 sampled pairs per batch, relu(0.01 - d), diag excluded.
// ---------------------------------------------------------------------------
__global__ void repulsion_kernel(const float* __restrict__ pred,
                                 const long long* __restrict__ sidx,
                                 int B, int N, int S, float* __restrict__ out) {
  __shared__ float s[256];
  const int total = B * S * S;
  float acc = 0.0f;
  for (int p = threadIdx.x; p < total; p += blockDim.x) {
    int b = p / (S * S);
    int rem = p % (S * S);
    int i = rem / S, j = rem % S;
    int ii = (int)sidx[i];
    int jj = (int)sidx[j];
    const float* A = pred + ((size_t)b * N + ii) * 3;
    const float* Bp = pred + ((size_t)b * N + jj) * 3;
    float dx = A[0] - Bp[0], dy = A[1] - Bp[1], dz = A[2] - Bp[2];
    float sq = dx * dx + dy * dy + dz * dz;
    float d = sq > 0.0f ? sqrtf(sq) : 0.0f;
    if (i == j) d += 1.0e10f;
    float v = 0.01f - d;
    acc += v > 0.0f ? v : 0.0f;
  }
  s[threadIdx.x] = acc;
  __syncthreads();
  for (int o = blockDim.x >> 1; o > 0; o >>= 1) {
    if (threadIdx.x < (unsigned)o) s[threadIdx.x] += s[threadIdx.x + o];
    __syncthreads();
  }
  if (threadIdx.x == 0) out[0] = s[0] * (5.0f / (float)total);
}

__global__ void finalize_kernel(const float* __restrict__ acc,
                                float* __restrict__ out) {
  out[0] = acc[0] + acc[1] + acc[2] + acc[3] + acc[4] + acc[5] + acc[6];
}

// ---------------------------------------------------------------------------
// Launch
// ---------------------------------------------------------------------------
extern "C" void kernel_launch(void* const* d_in, const int* in_sizes, int n_in,
                              void* d_out, int out_size, void* d_ws, size_t ws_size,
                              hipStream_t stream) {
  (void)in_sizes; (void)n_in; (void)out_size; (void)ws_size;

  const float*     pred    = (const float*)d_in[0];      // (2,4096,3) f32
  const float*     gt      = (const float*)d_in[1];      // (2,4096,3) f32
  const float*     partial = (const float*)d_in[2];      // (2,2048,3) f32
  const long long* sidx    = (const long long*)d_in[3];  // (100,) i64
  float*           out     = (float*)d_out;

  const int B = 2, N = 4096, M = 4096, K = 2048, S = 100;
  const int NF = B * N;  // 8192 flattened points

  float* ws      = (float*)d_ws;
  float* minPred = ws;                    // B*N
  float* minGt   = minPred + B * N;       // B*M
  float* minPart = minGt + B * M;         // B*K
  float* sortBuf = minPart + B * K;       // 6 * 8192 (pred cols 0..2, gt cols 0..2)
  float* rowStd  = sortBuf + 6 * NF;      // 8192
  float* acc     = rowStd + NF;           // 7 scalars

  const dim3 blk(256);

  // Chamfer (both directions) and coverage via WMMA pairwise-min
  pairwise_min_wmma<<<dim3(N / 16 / PMW_WAVES, B), blk, 0, stream>>>(pred, gt, N, M, minPred);
  pairwise_min_wmma<<<dim3(M / 16 / PMW_WAVES, B), blk, 0, stream>>>(gt, pred, M, N, minGt);
  pairwise_min_wmma<<<dim3(K / 16 / PMW_WAVES, B), blk, 0, stream>>>(partial, pred, K, N, minPart);

  reduce_sum_kernel<<<1, 1024, 0, stream>>>(minPred, B * N, 0, 1.0f / (B * N), acc + 0);
  reduce_sum_kernel<<<1, 1024, 0, stream>>>(minGt,   B * M, 0, 1.0f / (B * M), acc + 1);
  reduce_sum_kernel<<<1, 1024, 0, stream>>>(minPart, B * K, 1, 5.0f / (B * K), acc + 2);

  // EMD: sort each coordinate column of flattened pred / gt, then MSE * 0.1
  for (int c = 0; c < 3; ++c) {
    bitonic_sort_col<<<1, 1024, 0, stream>>>(pred, NF, c, sortBuf + c * NF);
    bitonic_sort_col<<<1, 1024, 0, stream>>>(gt,   NF, c, sortBuf + (3 + c) * NF);
  }
  emd_reduce_kernel<<<1, 1024, 0, stream>>>(sortBuf, NF, acc + 3);

  // Uniformity: per-row kNN(50) std, then mean * 2
  uniformity_rows_kernel<<<dim3(NF / 256), 256, 0, stream>>>(pred, NF, rowStd);
  reduce_sum_kernel<<<1, 1024, 0, stream>>>(rowStd, NF, 0, 2.0f / (float)NF, acc + 4);

  // Spread and repulsion
  spread_kernel<<<1, 32, 0, stream>>>(pred, B, N, acc + 5);
  repulsion_kernel<<<1, 256, 0, stream>>>(pred, sidx, B, N, S, acc + 6);

  finalize_kernel<<<1, 1, 0, stream>>>(acc, out);
}
